// LSTM_4123168604727
// MI455X (gfx1250) — matrix operand order repaired and checked by reading, not compiled, not run on hardware
//
#include <hip/hip_runtime.h>
#include <hip/hip_bf16.h>
#include <math.h>

// ---------------------------------------------------------------------------
// LSTM (Keras impl=2 style): xz = x@kernel + bias (big GEMM, parallel), then
// sequential scan over T with z = xz[t] + h@recurrent_kernel and gate math.
// BF16 WMMA (fp32 accumulate) on CDNA5 / gfx1250, wave32.
// Scan kernel holds its recurrent-weight WMMA fragments resident in VGPRs
// across all 1024 timesteps (128 VGPRs/wave), removing weight-load latency
// from the serialized dependency chain.
// ---------------------------------------------------------------------------

#define B_ 32
#define T_ 1024
#define D_ 512
#define U_ 512
#define G_ 2048   // 4*U

typedef __attribute__((ext_vector_type(16))) __bf16 v16bf;
typedef __attribute__((ext_vector_type(8)))  float  v8f;

union Frag { v16bf bf; unsigned u[8]; };
union Acc  { v8f f; float e[8]; };

__device__ __forceinline__ unsigned short f2bf(float f) {
    unsigned u = __builtin_bit_cast(unsigned, f);
    unsigned r = u + 0x7FFFu + ((u >> 16) & 1u);   // round-to-nearest-even
    return (unsigned short)(r >> 16);
}

__device__ __forceinline__ float hsig(float x) {
    return fminf(fmaxf(0.2f * x + 0.5f, 0.0f), 1.0f);
}

// ---------------- conversion kernels ----------------

__global__ __launch_bounds__(256)
void convert_flat(const float* __restrict__ src, unsigned short* __restrict__ dst, int n) {
    int i = blockIdx.x * blockDim.x + threadIdx.x;
    if (i < n) dst[i] = f2bf(src[i]);
}

// src: fp32 [rows][cols]  ->  dst: bf16 [cols][rows]  (K-contiguous for B frags)
__global__ __launch_bounds__(256)
void convert_tr(const float* __restrict__ src, unsigned short* __restrict__ dst,
                int rows, int cols) {
    int i = blockIdx.x * blockDim.x + threadIdx.x;
    if (i < rows * cols) {
        int r = i / cols, c = i % cols;
        dst[(size_t)c * rows + r] = f2bf(src[i]);
    }
}

__global__ void init_ctl(unsigned* ctl) {
    ctl[0] = 0u;   // arrive count must start at 0 (phase value is don't-care)
}

// ---------------- phase 1: xz = x @ kernel + bias ----------------
// One wave computes a 32x32 block (2x2 WMMA tiles). A from xbf [32768][512],
// B from wk (bf16, transposed [2048][512]). Output scattered to xz [T][B][G].

__global__ __launch_bounds__(256)
void xz_gemm(const unsigned short* __restrict__ xbf,
             const unsigned short* __restrict__ wk,
             const float* __restrict__ bias,
             float* __restrict__ xz) {
    const int lane = threadIdx.x & 31;
    const int wave = (blockIdx.x * blockDim.x + threadIdx.x) >> 5;
    const int wm = wave >> 6;          // 0..1023  (M blocks of 32 over 32768 rows)
    const int wn = wave & 63;          // 0..63    (N blocks of 32 over 2048 cols)
    const int m0 = wm * 32;
    const int n0 = wn * 32;
    const int r  = lane & 15;
    const int hi = lane >> 4;

    Acc acc[2][2];
    #pragma unroll
    for (int a = 0; a < 2; ++a)
        #pragma unroll
        for (int b = 0; b < 2; ++b)
            #pragma unroll
            for (int i = 0; i < 8; ++i) acc[a][b].e[i] = 0.0f;

    for (int k0 = 0; k0 < D_; k0 += 32) {
        Frag A[2], Bf[2];
        #pragma unroll
        for (int a = 0; a < 2; ++a) {
            const unsigned short* ap = xbf + (size_t)(m0 + a * 16 + r) * D_;
            #pragma unroll
            for (int j = 0; j < 4; ++j) {
                A[a].u[j]     = *(const unsigned*)(ap + k0 + hi * 8 + 2 * j);
                A[a].u[j + 4] = *(const unsigned*)(ap + k0 + 16 + hi * 8 + 2 * j);
            }
        }
        #pragma unroll
        for (int b = 0; b < 2; ++b) {
            const unsigned short* bp = wk + (size_t)(n0 + b * 16 + r) * D_;
            #pragma unroll
            for (int j = 0; j < 4; ++j) {
                Bf[b].u[j]     = *(const unsigned*)(bp + k0 + hi * 8 + 2 * j);
                Bf[b].u[j + 4] = *(const unsigned*)(bp + k0 + 16 + hi * 8 + 2 * j);
            }
        }
        #pragma unroll
        for (int a = 0; a < 2; ++a)
            #pragma unroll
            for (int b = 0; b < 2; ++b)
                acc[a][b].f = __builtin_amdgcn_wmma_f32_16x16x32_bf16(
                    false, A[a].bf, false, Bf[b].bf, (short)0, acc[a][b].f, false, false);
    }

    // epilogue: + bias, scatter to xz[t][b][g] (row m -> b = m>>10, t = m&1023)
    #pragma unroll
    for (int b = 0; b < 2; ++b) {
        const int g = n0 + b * 16 + r;
        const float bv = bias[g];
        #pragma unroll
        for (int a = 0; a < 2; ++a) {
            #pragma unroll
            for (int i = 0; i < 8; ++i) {
                const int m  = m0 + a * 16 + hi * 8 + i;
                const int t  = m & (T_ - 1);
                const int bb = m >> 10;
                xz[((size_t)t * B_ + bb) * G_ + g] = acc[a][b].e[i] + bv;
            }
        }
    }
}

// ---------------- phase 2: persistent recurrent scan ----------------

__device__ __forceinline__ void grid_barrier(unsigned* cnt, unsigned* phase, unsigned nblk) {
    __syncthreads();
    if (threadIdx.x == 0) {
        __threadfence();
        unsigned ph  = __hip_atomic_load(phase, __ATOMIC_ACQUIRE, __HIP_MEMORY_SCOPE_AGENT);
        unsigned old = __hip_atomic_fetch_add(cnt, 1u, __ATOMIC_ACQ_REL, __HIP_MEMORY_SCOPE_AGENT);
        if (old == nblk - 1u) {
            __hip_atomic_store(cnt, 0u, __ATOMIC_RELAXED, __HIP_MEMORY_SCOPE_AGENT);
            __hip_atomic_fetch_add(phase, 1u, __ATOMIC_RELEASE, __HIP_MEMORY_SCOPE_AGENT);
        } else {
            while (__hip_atomic_load(phase, __ATOMIC_ACQUIRE, __HIP_MEMORY_SCOPE_AGENT) == ph)
                __builtin_amdgcn_s_sleep(1);
        }
        __threadfence();
    }
    __syncthreads();
}

// 32 blocks x 256 threads = 256 waves; one wave per 16x16 tile of the 32x2048
// per-step GEMM. wr: bf16 [2048][512] (transposed recurrent kernel).
// Each wave's 16 B-fragments (its 16 output columns x K=512) stay in VGPRs
// for the whole scan: 16 frags x 8 VGPRs = 128 VGPRs of resident weights.
__global__ __launch_bounds__(256)
void lstm_scan(const float* __restrict__ xz,
               const unsigned short* __restrict__ wr,
               float* __restrict__ out,        // [B][T][U]
               unsigned short* hbf,            // [32][512] bf16 h state
               float* cbuf,                    // [32][512] fp32 c state
               float* zbuf,                    // [32][2048] fp32 z scratch
               unsigned* cnt, unsigned* phase) {
    const unsigned NBLK = 32;
    const int tid  = blockIdx.x * blockDim.x + threadIdx.x;  // 0..8191
    const int lane = threadIdx.x & 31;
    const int wave = tid >> 5;                               // 0..255
    const int mt   = wave & 1;                               // M tile (16 of the 32 rows)
    const int nt   = wave >> 1;                              // 0..127 N tile
    const int r    = lane & 15;
    const int hi   = lane >> 4;
    const int g    = nt * 16 + r;

    // init h = 0, c = 0
    for (int i = tid; i < B_ * U_; i += 8192) { hbf[i] = 0; cbuf[i] = 0.0f; }

    // Load ALL recurrent-weight fragments for this wave's columns into VGPRs.
    const unsigned short* bp = wr + (size_t)g * U_;   // column g, K-contiguous
    Frag Bfr[16];
    #pragma unroll
    for (int kk = 0; kk < 16; ++kk) {
        #pragma unroll
        for (int j = 0; j < 4; ++j) {
            Bfr[kk].u[j]     = *(const unsigned*)(bp + kk * 32 + hi * 8 + 2 * j);
            Bfr[kk].u[j + 4] = *(const unsigned*)(bp + kk * 32 + 16 + hi * 8 + 2 * j);
        }
    }

    grid_barrier(cnt, phase, NBLK);

    for (int t = 0; t < T_; ++t) {
        // z = xz[t] + h @ R   (accumulator seeded from xz -> free bias/input add)
        Acc acc;
        const float* xzp = xz + (size_t)t * B_ * G_;
        #pragma unroll
        for (int i = 0; i < 8; ++i) {
            const int bb = mt * 16 + hi * 8 + i;
            acc.e[i] = xzp[(size_t)bb * G_ + g];
        }

        const unsigned short* ap = hbf + (size_t)(mt * 16 + r) * U_;
        #pragma unroll
        for (int kk = 0; kk < 16; ++kk) {
            const int k0 = kk * 32;
            Frag A;
            #pragma unroll
            for (int j = 0; j < 4; ++j) {
                A.u[j]     = *(const unsigned*)(ap + k0 + hi * 8 + 2 * j);
                A.u[j + 4] = *(const unsigned*)(ap + k0 + 16 + hi * 8 + 2 * j);
            }
            acc.f = __builtin_amdgcn_wmma_f32_16x16x32_bf16(
                false, A.bf, false, Bfr[kk].bf, (short)0, acc.f, false, false);
        }

        #pragma unroll
        for (int i = 0; i < 8; ++i) {
            const int bb = mt * 16 + hi * 8 + i;
            zbuf[(size_t)bb * G_ + g] = acc.e[i];
        }

        grid_barrier(cnt, phase, NBLK);   // z complete

        // gate elementwise: 16384 elems over 8192 threads -> 2 each
        #pragma unroll
        for (int e = 0; e < 2; ++e) {
            const int idx = tid + e * 8192;
            const int bb  = idx >> 9;
            const int u   = idx & 511;
            const float* zr = zbuf + (size_t)bb * G_;
            const float zi = zr[u];
            const float zf = zr[512 + u];
            const float zc = zr[1024 + u];
            const float zo = zr[1536 + u];
            const float iv = hsig(zi);
            const float fv = hsig(zf);
            const float ov = hsig(zo);
            const float cn = fv * cbuf[idx] + iv * tanhf(zc);
            const float hn = ov * tanhf(cn);
            cbuf[idx] = cn;
            hbf[idx]  = f2bf(hn);
            out[((size_t)bb * T_ + t) * U_ + u] = hn;
        }

        grid_barrier(cnt, phase, NBLK);   // h ready for next step
    }
}

// ---------------- host launch ----------------

extern "C" void kernel_launch(void* const* d_in, const int* in_sizes, int n_in,
                              void* d_out, int out_size, void* d_ws, size_t ws_size,
                              hipStream_t stream) {
    const float* x    = (const float*)d_in[0];   // [B,T,D]
    const float* ker  = (const float*)d_in[1];   // [D,4U]
    const float* rker = (const float*)d_in[2];   // [U,4U]
    const float* bias = (const float*)d_in[3];   // [4U]
    float* out = (float*)d_out;                  // [B,T,U]

    char* ws = (char*)d_ws;
    size_t off = 0;
    auto carve = [&](size_t bytes) -> char* {
        off = (off + 255) & ~(size_t)255;
        char* p = ws + off;
        off += bytes;
        return p;
    };
    float*          xz   = (float*)         carve((size_t)T_ * B_ * G_ * 4);  // 256 MB
    unsigned short* xbf  = (unsigned short*)carve((size_t)B_ * T_ * D_ * 2);  //  32 MB
    unsigned short* wk   = (unsigned short*)carve((size_t)G_ * D_ * 2);       //   2 MB
    unsigned short* wr   = (unsigned short*)carve((size_t)G_ * U_ * 2);       //   2 MB
    float*          zbuf = (float*)         carve((size_t)B_ * G_ * 4);
    unsigned short* hbf  = (unsigned short*)carve((size_t)B_ * U_ * 2);
    float*          cbuf = (float*)         carve((size_t)B_ * U_ * 4);
    unsigned*       ctl  = (unsigned*)      carve(256);                       // [cnt, phase]

    // 1) precision conversion / weight transposition
    {
        int n = B_ * T_ * D_;
        convert_flat<<<(n + 255) / 256, 256, 0, stream>>>(x, xbf, n);
    }
    {
        int n = D_ * G_;
        convert_tr<<<(n + 255) / 256, 256, 0, stream>>>(ker, wk, D_, G_);
        convert_tr<<<(n + 255) / 256, 256, 0, stream>>>(rker, wr, U_, G_);
    }

    // 2) big input-projection GEMM: 65536 waves (1024 x 64 blocks of 32x32)
    xz_gemm<<<8192, 256, 0, stream>>>(xbf, wk, bias, xz);

    // 3) persistent recurrent scan (weights live in VGPRs for all 1024 steps)
    init_ctl<<<1, 1, 0, stream>>>(ctl);
    lstm_scan<<<32, 256, 0, stream>>>(xz, wr, out, hbf, cbuf, zbuf,
                                      ctl + 0, ctl + 1);
}
